// SelfAttention_81381040325094
// MI455X (gfx1250) — matrix operand order repaired and checked by reading, not compiled, not run on hardware
//
#include <hip/hip_runtime.h>
#include <hip/hip_bf16.h>

// ---------------------------------------------------------------------------
// Types / helpers
// ---------------------------------------------------------------------------
typedef __attribute__((ext_vector_type(8)))  __bf16 v8bf;
typedef __attribute__((ext_vector_type(16))) __bf16 v16bf;
typedef __attribute__((ext_vector_type(8)))  float  v8f;
typedef __attribute__((ext_vector_type(4)))  int    v4i;

static __device__ inline __bf16 f2bf(float f) {
  unsigned u = __builtin_bit_cast(unsigned, f);
  u += 0x7fffu + ((u >> 16) & 1u);               // round-to-nearest-even
  unsigned short h = (unsigned short)(u >> 16);
  return __builtin_bit_cast(__bf16, h);
}
static __device__ inline float bf2f(__bf16 b) {
  unsigned short h = __builtin_bit_cast(unsigned short, b);
  unsigned u = ((unsigned)h) << 16;
  return __builtin_bit_cast(float, u);
}
// Load a 16-element bf16 fragment as two contiguous 8-half (16B) chunks.
static __device__ inline v16bf ld_frag2(const __bf16* p0, const __bf16* p1) {
  v8bf a = *(const v8bf*)p0;
  v8bf b = *(const v8bf*)p1;
  return __builtin_shufflevector(a, b, 0,1,2,3,4,5,6,7,8,9,10,11,12,13,14,15);
}

// ---------------------------------------------------------------------------
// Async global -> LDS staging (gfx1250 ASYNCcnt path) with sync fallback.
// Probe-derived signature: (AS1 int4* src, AS3 int4* dst, imm offset, imm cpol).
// ---------------------------------------------------------------------------
#if __has_builtin(__builtin_amdgcn_global_load_async_to_lds_b128) && \
    __has_builtin(__builtin_amdgcn_s_wait_asynccnt)
#define ATHENA_ASYNC 1
typedef __attribute__((address_space(1))) v4i gv4i_t;
typedef __attribute__((address_space(3))) v4i lv4i_t;
#else
#define ATHENA_ASYNC 0
#endif

static __device__ inline void cp16(void* lds, const void* g) {
#if ATHENA_ASYNC
  __builtin_amdgcn_global_load_async_to_lds_b128((gv4i_t*)g, (lv4i_t*)lds, 0, 0);
#else
  *(v8bf*)lds = *(const v8bf*)g;
#endif
}
static __device__ inline void cp_wait() {
#if ATHENA_ASYNC
  __builtin_amdgcn_s_wait_asynccnt(0);
#endif
}

// ---------------------------------------------------------------------------
// fp32 -> bf16 conversion
// ---------------------------------------------------------------------------
__global__ __launch_bounds__(256) void k_f32_to_bf16(const float* __restrict__ in,
                                                     __bf16* __restrict__ out, long n) {
  long i = (long)blockIdx.x * blockDim.x + threadIdx.x;
  long stride = (long)gridDim.x * blockDim.x;
  for (; i < n; i += stride) out[i] = f2bf(in[i]);
}

// ---------------------------------------------------------------------------
// Generic bf16 WMMA GEMM:  C[M,N] = A[M,K] * B[N,K]^T
// Block tile 128x128, BK=32, 256 threads (8 waves), wave tile 32x64 (2x4 WMMA).
// Double-buffered LDS staging; tile i+1 streams in (async DMA when available)
// while tile i is consumed. One barrier per K-step.
// ---------------------------------------------------------------------------
template <bool OUT_BF16>
__global__ __launch_bounds__(256)
void k_gemm_bf16(const __bf16* __restrict__ A, const __bf16* __restrict__ B,
                 float* __restrict__ Cf, __bf16* __restrict__ Cb,
                 int M, int N, int K) {
  constexpr int BM = 128, BN = 128, BK = 32, LDT = BK + 8;   // padded LDS stride
  __shared__ __bf16 As[2][BM * LDT];
  __shared__ __bf16 Bs[2][BN * LDT];

  const int tid  = threadIdx.x;
  const int lane = tid & 31;
  const int wave = tid >> 5;
  const int lr = lane & 15, hi = lane >> 4;
  const int bm = blockIdx.y * BM;
  const int bn = blockIdx.x * BN;
  const int wm = (wave >> 1) * 32;    // wave row base (2 tiles)
  const int wn = (wave & 1) * 64;     // wave col base (4 tiles)

  v8f acc[2][4];
#pragma unroll
  for (int i = 0; i < 2; ++i)
#pragma unroll
    for (int j = 0; j < 4; ++j) acc[i][j] = (v8f)0.f;

  // Stage one 128x32 A tile + 128x32 B tile into LDS buffer `buf`.
  auto stage = [&](int buf, int k0) {
#pragma unroll
    for (int i = 0; i < 2; ++i) {
      int c  = tid + i * 256;
      int r  = c >> 2;              // 4 chunks of 8 halves per row
      int co = (c & 3) * 8;
      cp16(&As[buf][r * LDT + co], &A[(long)(bm + r) * K + k0 + co]);
      cp16(&Bs[buf][r * LDT + co], &B[(long)(bn + r) * K + k0 + co]);
    }
  };

  stage(0, 0);
  const int nk = K / BK;
  for (int it = 0; it < nk; ++it) {
    const int bsel = it & 1;
    cp_wait();                 // tile `it` landed in LDS (per-wave ASYNCcnt)
    __syncthreads();           // all waves see it; prior reads of buf^1 done
    if (it + 1 < nk) stage(bsel ^ 1, (it + 1) * BK);

    v16bf af[2], bfr[4];
#pragma unroll
    for (int i = 0; i < 2; ++i) {
      const __bf16* p = &As[bsel][(wm + i * 16 + lr) * LDT + hi * 8];
      af[i] = ld_frag2(p, p + 16);
    }
#pragma unroll
    for (int j = 0; j < 4; ++j) {
      const __bf16* p = &Bs[bsel][(wn + j * 16 + lr) * LDT + hi * 8];
      bfr[j] = ld_frag2(p, p + 16);
    }
#pragma unroll
    for (int i = 0; i < 2; ++i)
#pragma unroll
      for (int j = 0; j < 4; ++j)
        acc[i][j] = __builtin_amdgcn_wmma_f32_16x16x32_bf16(
            false, af[i], false, bfr[j], (short)0, acc[i][j], false, false);
  }

  // Epilogue: C layout — VGPR e holds row e (lanes 0-15) / e+8 (lanes 16-31).
#pragma unroll
  for (int i = 0; i < 2; ++i)
#pragma unroll
    for (int j = 0; j < 4; ++j)
#pragma unroll
      for (int e = 0; e < 8; ++e) {
        long row = bm + wm + i * 16 + hi * 8 + e;
        long col = bn + wn + j * 16 + lr;
        if (OUT_BF16) Cb[row * N + col] = f2bf(acc[i][j][e]);
        else          Cf[row * N + col] = acc[i][j][e];
      }
}

// ---------------------------------------------------------------------------
// RoPE in place on fused QKV bf16 buffer [B*S, 6144]; heads 0..31 = Q, 32..39 = K.
// ---------------------------------------------------------------------------
__global__ __launch_bounds__(64)
void k_rope(__bf16* __restrict__ qkv, const float* __restrict__ fcos,
            const float* __restrict__ fsin) {
  int tok = blockIdx.x;                 // 0..4095 (b*2048 + s)
  int hy  = blockIdx.y;                 // 0..39
  int i   = threadIdx.x;                // 0..63 (pair index)
  int s   = tok & 2047;
  long base = (long)tok * 6144 + (hy < 32 ? hy * 128 : 4096 + (hy - 32) * 128);
  float c  = fcos[s * 64 + i];
  float sn = fsin[s * 64 + i];
  float x0 = bf2f(qkv[base + 2 * i]);
  float x1 = bf2f(qkv[base + 2 * i + 1]);
  qkv[base + 2 * i]     = f2bf(x0 * c - x1 * sn);
  qkv[base + 2 * i + 1] = f2bf(x0 * sn + x1 * c);
}

// ---------------------------------------------------------------------------
// Flash attention (non-causal, start_pos=0).
// One block per (b, head, 64-row Q tile). 256 threads = 8 waves.
// Online softmax: thread r (<64) owns score row r across all key chunks.
// K/V chunks staged through the async-to-LDS path when available.
// ---------------------------------------------------------------------------
__global__ __launch_bounds__(256)
void k_attn(const __bf16* __restrict__ qkv, __bf16* __restrict__ attn_out) {
  constexpr int S = 2048, HD = 128, ROW = 6144, CH = 64, QT = 64;
  constexpr int LQ = HD + 8;   // bf16 stride 136 halves (16B aligned rows)
  constexpr int LS = 68;       // f32 stride
  constexpr int LP = CH + 8;   // bf16 stride 72

  __shared__ __bf16 Qs[QT * LQ];
  __shared__ __bf16 Ks[CH * LQ];
  __shared__ __bf16 Vs[CH * LQ];
  __shared__ float  Ss[QT * LS];
  __shared__ __bf16 Ps[QT * LP];
  __shared__ float  row_scale[QT];
  __shared__ float  row_inv[QT];

  const int tid  = threadIdx.x;
  const int lane = tid & 31;
  const int wave = tid >> 5;
  const int lr = lane & 15, hi = lane >> 4;

  const int blk = blockIdx.x;
  const int qt  = blk & 31;
  const int h   = (blk >> 5) & 31;
  const int b   = blk >> 10;
  const int g   = h >> 2;                      // kv head (n_rep = 4)

  const long base_tok = (long)b * S * ROW;
  const __bf16* Qg = qkv + base_tok + (long)h * HD;
  const __bf16* Kg = qkv + base_tok + 4096 + (long)g * HD;
  const __bf16* Vg = qkv + base_tok + 4096 + 1024 + (long)g * HD;

  // Load 64x128 Q tile (4 x 16B chunks / thread).
#pragma unroll
  for (int i = 0; i < 4; ++i) {
    int c = tid + i * 256;
    int r = c >> 4;                 // 16 chunks of 8 halves per row
    int co = (c & 15) * 8;
    cp16(&Qs[r * LQ + co], &Qg[(long)(qt * QT + r) * ROW + co]);
  }

  const int mo  = wave >> 1;          // row-tile index 0..3
  const int nt0 = (wave & 1) * 2;     // S column tiles (2 per wave)
  const int no0 = (wave & 1) * 4;     // O column tiles (4 per wave)

  v8f oacc[4];
#pragma unroll
  for (int j = 0; j < 4; ++j) oacc[j] = (v8f)0.f;

  float m_r = -3.0e38f, l_r = 0.f;    // live in threads 0..63 only

  for (int kc = 0; kc < S / CH; ++kc) {
    __syncthreads();   // previous P*V reads done before overwriting Ks/Vs
#pragma unroll
    for (int i = 0; i < 4; ++i) {
      int c = tid + i * 256;
      int r = c >> 4;
      int co = (c & 15) * 8;
      cp16(&Ks[r * LQ + co], &Kg[(long)(kc * CH + r) * ROW + co]);
      cp16(&Vs[r * LQ + co], &Vg[(long)(kc * CH + r) * ROW + co]);
    }
    cp_wait();          // K/V (and, on kc==0, the Q tile) landed in LDS
    __syncthreads();

    // ---- S = (Q K^T) * 1/sqrt(128) : each wave 2 tiles of 16x16 ----
#pragma unroll
    for (int t = 0; t < 2; ++t) {
      int nt = nt0 + t;
      v8f sacc = (v8f)0.f;
#pragma unroll
      for (int k0 = 0; k0 < HD; k0 += 32) {
        const __bf16* pa = &Qs[(mo * 16 + lr) * LQ + k0 + hi * 8];
        const __bf16* pb = &Ks[(nt * 16 + lr) * LQ + k0 + hi * 8];
        v16bf af  = ld_frag2(pa, pa + 16);
        v16bf bfr = ld_frag2(pb, pb + 16);
        sacc = __builtin_amdgcn_wmma_f32_16x16x32_bf16(
            false, af, false, bfr, (short)0, sacc, false, false);
      }
#pragma unroll
      for (int e = 0; e < 8; ++e)
        Ss[(mo * 16 + hi * 8 + e) * LS + nt * 16 + lr] = sacc[e] * 0.08838834764831845f;
    }
    __syncthreads();

    // ---- online softmax per row ----
    if (tid < QT) {
      int r = tid;
      float cm = -3.0e38f;
#pragma unroll 8
      for (int c = 0; c < CH; ++c) cm = fmaxf(cm, Ss[r * LS + c]);
      float mn = fmaxf(m_r, cm);
      float sc = __expf(m_r - mn);
      float sum = 0.f;
#pragma unroll 8
      for (int c = 0; c < CH; ++c) {
        float p = __expf(Ss[r * LS + c] - mn);
        sum += p;
        Ps[r * LP + c] = f2bf(p);
      }
      l_r = l_r * sc + sum;
      m_r = mn;
      row_scale[r] = sc;
    }
    __syncthreads();

    // ---- rescale O, then O += P * V ----
    float fs[8];
#pragma unroll
    for (int e = 0; e < 8; ++e) fs[e] = row_scale[mo * 16 + hi * 8 + e];
#pragma unroll
    for (int j = 0; j < 4; ++j)
#pragma unroll
      for (int e = 0; e < 8; ++e) oacc[j][e] *= fs[e];

#pragma unroll
    for (int j = 0; j < 4; ++j) {
      int no = no0 + j;
#pragma unroll
      for (int k0 = 0; k0 < CH; k0 += 32) {
        const __bf16* pa = &Ps[(mo * 16 + lr) * LP + k0 + hi * 8];
        v16bf af = ld_frag2(pa, pa + 16);
        // B fragment: lane owns column d = no*16 + lr, k in table order (k0 + hi*8 + s, +16 for s>=8).
        const __bf16* pb = &Vs[(k0 + hi * 8) * LQ + no * 16 + lr];
        v16bf bfr;
#pragma unroll
        for (int s = 0; s < 16; ++s) bfr[s] = pb[((s & 7) + (s >> 3) * 16) * LQ];
        oacc[j] = __builtin_amdgcn_wmma_f32_16x16x32_bf16(
            false, af, false, bfr, (short)0, oacc[j], false, false);
      }
    }
  }

  __syncthreads();
  if (tid < QT) row_inv[tid] = 1.0f / l_r;
  __syncthreads();

  float inv[8];
#pragma unroll
  for (int e = 0; e < 8; ++e) inv[e] = row_inv[mo * 16 + hi * 8 + e];
#pragma unroll
  for (int j = 0; j < 4; ++j) {
    int no = no0 + j;
#pragma unroll
    for (int e = 0; e < 8; ++e) {
      long srow = (long)b * S + qt * QT + mo * 16 + hi * 8 + e;
      attn_out[srow * 4096 + h * 128 + no * 16 + lr] = f2bf(oacc[j][e] * inv[e]);
    }
  }
}

// ---------------------------------------------------------------------------
// Host-side launch
// ---------------------------------------------------------------------------
extern "C" void kernel_launch(void* const* d_in, const int* in_sizes, int n_in,
                              void* d_out, int out_size, void* d_ws, size_t ws_size,
                              hipStream_t stream) {
  const float* x    = (const float*)d_in[0];
  const float* wq   = (const float*)d_in[1];
  const float* wk   = (const float*)d_in[2];
  const float* wv   = (const float*)d_in[3];
  const float* wo   = (const float*)d_in[4];
  const float* fcos = (const float*)d_in[7];
  const float* fsin = (const float*)d_in[8];
  float* out = (float*)d_out;

  const long BS = 2L * 2048;          // tokens
  const long D  = 4096;
  const long NQKV = 6144;             // 4096 q + 1024 k + 1024 v

  char* ws = (char*)d_ws;
  size_t off = 0;
  auto take = [&](size_t elems) {
    void* p = ws + off;
    off = (off + elems * sizeof(__bf16) + 255) & ~(size_t)255;
    return (__bf16*)p;
  };
  __bf16* x_bf  = take(BS * D);         // 32 MiB
  __bf16* wqkv  = take(NQKV * D);       // 48 MiB (wq rows 0..4095, wk 4096..5119, wv 5120..6143)
  __bf16* wo_bf = take(D * D);          // 32 MiB
  __bf16* qkv   = take(BS * NQKV);      // 48 MiB
  __bf16* attn  = take(BS * D);         // 32 MiB

  // 1) fp32 -> bf16 conversions
  k_f32_to_bf16<<<2048, 256, 0, stream>>>(x,  x_bf,             BS * D);
  k_f32_to_bf16<<<2048, 256, 0, stream>>>(wq, wqkv,             D * D);
  k_f32_to_bf16<<<1024, 256, 0, stream>>>(wk, wqkv + D * D,     1024L * D);
  k_f32_to_bf16<<<1024, 256, 0, stream>>>(wv, wqkv + 5120L * D, 1024L * D);
  k_f32_to_bf16<<<2048, 256, 0, stream>>>(wo, wo_bf,            D * D);

  // 2) fused QKV projection: [4096 x 4096] * [6144 x 4096]^T -> bf16 [4096 x 6144]
  dim3 gq((unsigned)(NQKV / 128), (unsigned)(BS / 128));
  k_gemm_bf16<true><<<gq, 256, 0, stream>>>(x_bf, wqkv, nullptr, qkv,
                                            (int)BS, (int)NQKV, (int)D);

  // 3) RoPE on Q (32 heads) and K (8 heads), start_pos = 0
  dim3 gr((unsigned)BS, 40);
  k_rope<<<gr, 64, 0, stream>>>(qkv, fcos, fsin);

  // 4) flash attention: B * 32 heads * 32 q-tiles = 2048 blocks
  k_attn<<<2048, 256, 0, stream>>>(qkv, attn);

  // 5) output projection: attn [4096 x 4096] * wo [4096 x 4096]^T -> fp32 out
  dim3 go((unsigned)(D / 128), (unsigned)(BS / 128));
  k_gemm_bf16<false><<<go, 256, 0, stream>>>(attn, wo_bf, out, nullptr,
                                             (int)BS, (int)D, (int)D);
}
